// DetectionTargetLayer_20512763806373
// MI455X (gfx1250) — matrix operand order repaired and checked by reading, not compiled, not run on hardware
//
#include <hip/hip_runtime.h>

#define NPROP 2000
#define NGT   100
#define TROIS 200
#define PMAX  66
#define MS    28
#define IMGW  512
#define WS_B  1200   // per-batch workspace stride in ints: assign[200] isp[200] proi[200*4]

typedef __attribute__((ext_vector_type(16))) _Float16 v16h;
typedef __attribute__((ext_vector_type(8)))  float    v8f;

// ---------------------------------------------------------------- zero output
__global__ void k_zero(float* __restrict__ out, int n) {
  int i = blockIdx.x * 256 + threadIdx.x;
  if (i < n) out[i] = 0.0f;
}

// ---------------------------------------------------- selection + rois/cls/deltas
// one block of 1024 threads per batch item
__global__ void k_select(const float* __restrict__ proposals,
                         const int*   __restrict__ gt_cls,
                         const float* __restrict__ gt_boxes,
                         float* __restrict__ out,
                         int*   __restrict__ ws,
                         int off_cls, int off_del) {
  const int b = blockIdx.x;
  const int t = threadIdx.x;

  __shared__ float gtb[NGT * 4];
  __shared__ int   fgf[NGT];
  __shared__ int   crf[NGT];
  __shared__ int   scan[1024];
  __shared__ int   pord[NPROP];
  __shared__ int   nord[NPROP];
  __shared__ int   cnts[4];   // Tpos, Tneg, num_pos, num_neg

  const float* prop = proposals + b * NPROP * 4;
  const float* gbx  = gt_boxes  + b * NGT * 4;
  const int*   gcl  = gt_cls    + b * NGT;

  if (t < NGT) {
    float g0 = gbx[t*4+0], g1 = gbx[t*4+1], g2 = gbx[t*4+2], g3 = gbx[t*4+3];
    gtb[t*4+0] = g0; gtb[t*4+1] = g1; gtb[t*4+2] = g2; gtb[t*4+3] = g3;
    int valid = (fabsf(g0) + fabsf(g1) + fabsf(g2) + fabsf(g3)) > 0.0f;
    int c = gcl[t];
    fgf[t] = valid && (c > 0);
    crf[t] = valid && (c < 0);
  }
  __syncthreads();

  // --- per-proposal pos/neg flags (2 proposals per thread, original order kept)
  int pflag[2] = {0, 0}, nflag[2] = {0, 0};
  for (int e = 0; e < 2; ++e) {
    int i = 2 * t + e;
    if (i < NPROP) {
      float p0 = prop[i*4+0], p1 = prop[i*4+1], p2 = prop[i*4+2], p3 = prop[i*4+3];
      int valid = (fabsf(p0) + fabsf(p1) + fabsf(p2) + fabsf(p3)) > 0.0f;
      float fgmax = 0.0f, crmax = 0.0f;
      if (valid) {
        float a1 = (p2 - p0) * (p3 - p1);
        for (int g = 0; g < NGT; ++g) {
          float g0 = gtb[g*4+0], g1 = gtb[g*4+1], g2 = gtb[g*4+2], g3 = gtb[g*4+3];
          float yy1 = fmaxf(p0, g0), xx1 = fmaxf(p1, g1);
          float yy2 = fminf(p2, g2), xx2 = fminf(p3, g3);
          float inter = fmaxf(xx2 - xx1, 0.0f) * fmaxf(yy2 - yy1, 0.0f);
          float a2 = (g2 - g0) * (g3 - g1);
          float un = a1 + a2 - inter;
          float iou = (un > 0.0f) ? inter / un : 0.0f;
          if (fgf[g]) fgmax = fmaxf(fgmax, iou);
          if (crf[g]) crmax = fmaxf(crmax, iou);
        }
      }
      pflag[e] = valid && (fgmax >= 0.5f);
      nflag[e] = valid && (fgmax < 0.5f) && (crmax < 0.001f);
    }
  }

  // --- packed inclusive scan over 1024 per-thread sums (pos low16, neg high16)
  int psum = pflag[0] + pflag[1];
  int nsum = nflag[0] + nflag[1];
  scan[t] = psum | (nsum << 16);
  __syncthreads();
  for (int off = 1; off < 1024; off <<= 1) {
    int v = (t >= off) ? scan[t - off] : 0;
    __syncthreads();
    scan[t] += v;
    __syncthreads();
  }
  int incl = scan[t];
  int tot  = scan[1023];
  int Tpos = tot & 0xffff;
  int Tneg = tot >> 16;
  int ep = (incl & 0xffff) - psum;  // exclusive pos-count before element 2t
  int en = (incl >> 16)    - nsum;

  // --- stable partitions == stable argsort(!flag)
  for (int e = 0; e < 2; ++e) {
    int i = 2 * t + e;
    if (i < NPROP) {
      if (pflag[e]) pord[ep] = i; else pord[Tpos + (i - ep)] = i;
      if (nflag[e]) nord[en] = i; else nord[Tneg + (i - en)] = i;
      ep += pflag[e];
      en += nflag[e];
    }
  }
  if (t == 0) {
    int np = (Tpos < PMAX) ? Tpos : PMAX;
    int want = (int)floorf((float)(1.0 / 0.33) * (float)np) - np;
    int nn = (want < Tneg) ? want : Tneg;
    if (nn < 0) nn = 0;
    if (nn > TROIS - np) nn = TROIS - np;
    cnts[0] = Tpos; cnts[1] = Tneg; cnts[2] = np; cnts[3] = nn;
  }
  __syncthreads();

  // --- per output slot j
  if (t < TROIS) {
    int j = t;
    int np = cnts[2], nn = cnts[3];
    int is_p = (j < np);
    int is_n = (j >= np) && (j < np + nn);
    int idxn = j - np; if (idxn < 0) idxn = 0;
    int sel = is_p ? pord[j] : nord[idxn];
    float keep = (is_p || is_n) ? 1.0f : 0.0f;

    float p0 = prop[sel*4+0], p1 = prop[sel*4+1], p2 = prop[sel*4+2], p3 = prop[sel*4+3];
    float r0 = p0 * keep, r1 = p1 * keep, r2 = p2 * keep, r3 = p3 * keep;
    out[(b * TROIS + j) * 4 + 0] = r0;
    out[(b * TROIS + j) * 4 + 1] = r1;
    out[(b * TROIS + j) * 4 + 2] = r2;
    out[(b * TROIS + j) * 4 + 3] = r3;

    // assign = argmax_g where(is_fg, iou(sel,g), -1)  (first max)
    int validp = (fabsf(p0) + fabsf(p1) + fabsf(p2) + fabsf(p3)) > 0.0f;
    float a1 = (p2 - p0) * (p3 - p1);
    float best = -3.402823466e38f;
    int bi = 0;
    for (int g = 0; g < NGT; ++g) {
      float g0 = gtb[g*4+0], g1 = gtb[g*4+1], g2 = gtb[g*4+2], g3 = gtb[g*4+3];
      float iou = 0.0f;
      if (validp) {
        float yy1 = fmaxf(p0, g0), xx1 = fmaxf(p1, g1);
        float yy2 = fminf(p2, g2), xx2 = fminf(p3, g3);
        float inter = fmaxf(xx2 - xx1, 0.0f) * fmaxf(yy2 - yy1, 0.0f);
        float a2 = (g2 - g0) * (g3 - g1);
        float un = a1 + a2 - inter;
        iou = (un > 0.0f) ? inter / un : 0.0f;
      }
      float m = fgf[g] ? iou : -1.0f;
      if (m > best) { best = m; bi = g; }
    }

    int cls = is_p ? gcl[bi] : 0;
    out[off_cls + b * TROIS + j] = (float)cls;

    float q0, q1, q2, q3, G0, G1, G2, G3;
    if (is_p) { q0 = r0; q1 = r1; q2 = r2; q3 = r3; }
    else      { q0 = 0.0f; q1 = 0.0f; q2 = 1.0f; q3 = 1.0f; }
    if (is_p) { G0 = gtb[bi*4+0]; G1 = gtb[bi*4+1]; G2 = gtb[bi*4+2]; G3 = gtb[bi*4+3]; }
    else      { G0 = 0.0f; G1 = 0.0f; G2 = 1.0f; G3 = 1.0f; }

    float h  = q2 - q0,  w  = q3 - q1;
    float cy = q0 + 0.5f * h, cx = q1 + 0.5f * w;
    float gh = G2 - G0, gw = G3 - G1;
    float gcy = G0 + 0.5f * gh, gcx = G1 + 0.5f * gw;
    float fp = is_p ? 1.0f : 0.0f;
    out[off_del + (b * TROIS + j) * 4 + 0] = (((gcy - cy) / h) / 0.1f) * fp;
    out[off_del + (b * TROIS + j) * 4 + 1] = (((gcx - cx) / w) / 0.1f) * fp;
    out[off_del + (b * TROIS + j) * 4 + 2] = (logf(gh / h) / 0.2f) * fp;
    out[off_del + (b * TROIS + j) * 4 + 3] = (logf(gw / w) / 0.2f) * fp;

    int base = b * WS_B;
    ws[base + j]       = bi;
    ws[base + 200 + j] = is_p;
    float* wsf = (float*)ws;
    wsf[base + 400 + j*4 + 0] = q0;
    wsf[base + 400 + j*4 + 1] = q1;
    wsf[base + 400 + j*4 + 2] = q2;
    wsf[base + 400 + j*4 + 3] = q3;
  }
}

// ------------------------------------------- mask crop+resize via WMMA GEMM
// grid (PMAX, B), 256 threads = 8 waves; per ROI: OUT28x28 = Wy * (IMG * Wx^T)
// horizontal pass as (64x512)x(512x32) f16 WMMA GEMM, K chunked by 128 in LDS.
__global__ void k_masks(const int* __restrict__ gmask,
                        const int* __restrict__ ws,
                        float* __restrict__ out,
                        int off_masks) {
  const int p   = blockIdx.x;
  const int b   = blockIdx.y;
  const int tid = threadIdx.x;

  __shared__ int   s_isp, s_g;
  __shared__ float s_box[4];
  __shared__ int   y0i[MS], y1i[MS], x0i[MS], x1i[MS], oky[MS], okx[MS];
  __shared__ float wys[MS], wxs[MS];
  __shared__ _Float16 Rl[64 * 128];   // image-row chunk, f16
  __shared__ float    Tl[64 * 32];    // horizontal-interp result

  const int base = b * WS_B;
  if (tid == 0) {
    s_isp = ws[base + 200 + p];
    s_g   = ws[base + p];
    const float* wsf = (const float*)ws;
    s_box[0] = wsf[base + 400 + p*4 + 0];
    s_box[1] = wsf[base + 400 + p*4 + 1];
    s_box[2] = wsf[base + 400 + p*4 + 2];
    s_box[3] = wsf[base + 400 + p*4 + 3];
  }
  __syncthreads();
  if (!s_isp) return;   // uniform: output region already zeroed

  if (tid < MS) {
    int r = tid;
    float yA = s_box[0], yB = s_box[2];
    float scy = (yB - yA) * 511.0f / 27.0f;
    float ysv = yA * 511.0f + (float)r * scy;
    float yf = floorf(ysv);
    wys[r] = ysv - yf;
    y0i[r] = (int)fminf(fmaxf(yf, 0.0f), 511.0f);
    y1i[r] = (int)fminf(fmaxf(yf + 1.0f, 0.0f), 511.0f);
    oky[r] = (ysv >= 0.0f) && (ysv <= 511.0f);

    float xA = s_box[1], xB = s_box[3];
    float scx = (xB - xA) * 511.0f / 27.0f;
    float xsv = xA * 511.0f + (float)r * scx;
    float xf = floorf(xsv);
    wxs[r] = xsv - xf;
    x0i[r] = (int)fminf(fmaxf(xf, 0.0f), 511.0f);
    x1i[r] = (int)fminf(fmaxf(xf + 1.0f, 0.0f), 511.0f);
    okx[r] = (xsv >= 0.0f) && (xsv <= 511.0f);
  }
  __syncthreads();

  const int lane = tid & 31;
  const int wv   = tid >> 5;
  const int mt   = wv >> 1;           // 4 M-tiles (rows 0..63)
  const int nt   = wv & 1;            // 2 N-tiles (cols 0..31)
  const int lm   = lane & 15;
  const int hiA  = (lane >> 4) * 8;   // A K-base per lane half
  const int hiB  = (lane >> 4) * 16;  // unused; B also takes hiA as its K-base
  const int rowA = mt * 16 + lm;      // R row this lane reads
  const int colB = nt * 16 + lm;      // output column this lane owns

  float wxc = (colB < MS) ? wxs[colB] : 0.0f;
  int   cx0 = (colB < MS) ? x0i[colB] : -1;
  int   cx1 = (colB < MS) ? x1i[colB] : -1;

  const int* img = gmask + b * (IMGW * IMGW * NGT) + s_g;  // element (y*512+x)*100

  v8f acc = {0.0f, 0.0f, 0.0f, 0.0f, 0.0f, 0.0f, 0.0f, 0.0f};

  for (int ck = 0; ck < 4; ++ck) {
    // gather 64x128 f16 chunk of needed image rows (rows 56..63 zero)
    for (int idx = tid; idx < 64 * 128; idx += 256) {
      int row = idx >> 7;
      int xk  = idx & 127;
      _Float16 hv = (_Float16)0.0f;
      if (row < 56) {
        int rr = row >> 1;
        int y  = (row & 1) ? y1i[rr] : y0i[rr];
        int x  = ck * 128 + xk;
        int v  = img[(y * IMGW + x) * NGT];
        hv = v ? (_Float16)1.0f : (_Float16)0.0f;
      }
      Rl[idx] = hv;
    }
    __syncthreads();

    for (int ks = 0; ks < 4; ++ks) {
      v16h a, bf;
      int kg = ck * 128 + ks * 32;
#pragma unroll
      for (int h = 0; h < 16; ++h) {
        int v  = h >> 1;
        int lo = h & 1;
        // A (16x32 f16): K = [16 if v>=4] + hiA + 2*(v&3) + lo
        int kA = ((v & 4) ? 16 : 0) + hiA + 2 * (v & 3) + lo;
        a[h] = Rl[rowA * 128 + ks * 32 + kA];
        // B (32x16 f16): K = [16 if h>=8] + hiA + (h&7); sparse Wx synthesized in-register
        int kB = kg + ((h & 8) ? 16 : 0) + hiA + (h & 7);
        float bw = 0.0f;
        if (kB == cx0) bw += 1.0f - wxc;
        if (kB == cx1) bw += wxc;
        bf[h] = (_Float16)bw;
      }
      acc = __builtin_amdgcn_wmma_f32_16x16x32_f16(false, a, false, bf,
                                                   (short)0, acc, false, false);
    }
    __syncthreads();
  }

  // spill D (C/D layout: VGPR v -> M = v + 8*(lane>=16), N = lane&15)
  {
    int mb = (lane >> 4) * 8;
#pragma unroll
    for (int v = 0; v < 8; ++v) {
      Tl[(mt * 16 + mb + v) * 32 + nt * 16 + lm] = acc[v];
    }
  }
  __syncthreads();

  // vertical 2-tap combine + bounds mask + round-half-even
  for (int idx = tid; idx < MS * MS; idx += 256) {
    int r = idx / MS;
    int c = idx - r * MS;
    float wy = wys[r];
    float t0 = Tl[(2 * r) * 32 + c];
    float t1 = Tl[(2 * r + 1) * 32 + c];
    float val = t0 * (1.0f - wy) + t1 * wy;
    if (!(oky[r] && okx[c])) val = 0.0f;
    out[off_masks + ((b * TROIS + p) * MS + r) * MS + c] = rintf(val);
  }
}

// ---------------------------------------------------------------------- launch
extern "C" void kernel_launch(void* const* d_in, const int* in_sizes, int n_in,
                              void* d_out, int out_size, void* d_ws, size_t ws_size,
                              hipStream_t stream) {
  const float* proposals = (const float*)d_in[0];
  const int*   gt_cls    = (const int*)  d_in[1];
  const float* gt_boxes  = (const float*)d_in[2];
  const int*   gt_masks  = (const int*)  d_in[3];
  float* out = (float*)d_out;
  int*   ws  = (int*)d_ws;

  const int B = in_sizes[0] / (NPROP * 4);
  const int off_cls   = B * TROIS * 4;
  const int off_del   = off_cls + B * TROIS;
  const int off_masks = off_del + B * TROIS * 4;

  k_zero<<<(out_size + 255) / 256, 256, 0, stream>>>(out, out_size);
  k_select<<<B, 1024, 0, stream>>>(proposals, gt_cls, gt_boxes, out, ws,
                                   off_cls, off_del);
  k_masks<<<dim3(PMAX, B), 256, 0, stream>>>(gt_masks, ws, out, off_masks);
}
